// Model_1_54013508714660
// MI455X (gfx1250) — compile-verified, matrix operand order checked by reading, not run on hardware
//
#include <hip/hip_runtime.h>
#include <hip/hip_bf16.h>
#include <hip/hip_fp16.h>

typedef _Float16 f16;
typedef __attribute__((ext_vector_type(16))) _Float16 v16h;
typedef __attribute__((ext_vector_type(8)))  float    v8f;

// Async global->LDS staging path (gfx1250). Guarded so compile never breaks.
#if defined(__AMDGCN__) && __has_builtin(__builtin_amdgcn_global_load_async_to_lds_b128) && \
    __has_builtin(__builtin_amdgcn_s_wait_asynccnt)
#define USE_ASYNC_LDS 1
// Builtin signature (from probe diagnostics): param0 = v4i (vector_size 16) in
// device/global AS, param1 = LDS AS, params 2/3 = imm offset / imm cpol.
typedef int v4i_gcc __attribute__((vector_size(16)));
typedef __attribute__((address_space(1))) v4i_gcc* as1_v4i;
typedef __attribute__((address_space(3))) v4i_gcc* as3_v4i;
#else
#define USE_ASYNC_LDS 0
#endif

// ---------------------------------------------------------------------------
// WMMA helpers (CDNA5 gfx1250, wave32). D = A(16x32 f16) x B(32x16 f16) + C(f32)
// ---------------------------------------------------------------------------
__device__ __forceinline__ v8f wmma32(v16h a, v16h b, v8f c) {
  return __builtin_amdgcn_wmma_f32_16x16x32_f16(
      /*neg_a=*/false, a, /*neg_b=*/false, b,
      /*c_mod=*/(short)0, c, /*reuse_a=*/false, /*reuse_b=*/false);
}

// 16-bit A (16x32) lane layout (ISA 7.12.2): lane m=lane&15, hi=lane>>4;
// VGPR v in 0..3 -> K = v*2 + hi*8 + {0,1}; v in 4..7 -> K = 16 + (v-4)*2 + hi*8.
// Per lane this is two contiguous 16B chunks -> compiler emits global_load_b128.
__device__ __forceinline__ v16h ld_tile(const f16* __restrict__ base, int ld, int lane) {
  const int m  = lane & 15;
  const int hi = (lane >> 4) & 1;
  const f16* r = base + (size_t)m * ld;
  v16h out;
#pragma unroll
  for (int v = 0; v < 8; ++v) {
    const int k = ((v & 4) ? 16 : 0) + (v & 3) * 2 + hi * 8;
    out[2 * v]     = r[k];
    out[2 * v + 1] = r[k + 1];
  }
  return out;
}

// same, but zero-pad K >= kmax (used when HEAD_D=8 is padded to K=32)
__device__ __forceinline__ v16h ld_tile_pad(const f16* __restrict__ base, int ld,
                                            int lane, int kmax) {
  const int m  = lane & 15;
  const int hi = (lane >> 4) & 1;
  const f16* r = base + (size_t)m * ld;
  v16h out;
#pragma unroll
  for (int v = 0; v < 8; ++v) {
    const int k = ((v & 4) ? 16 : 0) + (v & 3) * 2 + hi * 8;
    out[2 * v]     = (k     < kmax) ? r[k]     : (f16)0.0f;
    out[2 * v + 1] = (k + 1 < kmax) ? r[k + 1] : (f16)0.0f;
  }
  return out;
}

// load A fragment from an f32 source (softmax probabilities in LDS)
__device__ __forceinline__ v16h ld_tile_f32(const float* __restrict__ base, int ld, int lane) {
  const int m  = lane & 15;
  const int hi = (lane >> 4) & 1;
  const float* r = base + (size_t)m * ld;
  v16h out;
#pragma unroll
  for (int v = 0; v < 8; ++v) {
    const int k = ((v & 4) ? 16 : 0) + (v & 3) * 2 + hi * 8;
    out[2 * v]     = (f16)r[k];
    out[2 * v + 1] = (f16)r[k + 1];
  }
  return out;
}

// B fragment from V (row-major SxHD, HD=8; columns n>=8 are zero padding)
__device__ __forceinline__ v16h ld_tileB_v(const f16* __restrict__ V, int lane, int k0) {
  const int n  = lane & 15;
  const int hi = (lane >> 4) & 1;
  v16h out;
#pragma unroll
  for (int v = 0; v < 8; ++v) {
    const int k = k0 + ((v & 4) ? 16 : 0) + (v & 3) * 2 + hi * 8;
    out[2 * v]     = (n < 8) ? V[k * 8 + n]       : (f16)0.0f;
    out[2 * v + 1] = (n < 8) ? V[(k + 1) * 8 + n] : (f16)0.0f;
  }
  return out;
}

// C/D layout (ISA 7.12.2): element (M,N): lane = N + 16*(M>=8), VGPR = M%8
__device__ __forceinline__ void st_cd(float* __restrict__ base, int ld, int lane, v8f c) {
  const int n  = lane & 15;
  const int hi = (lane >> 4) & 1;
#pragma unroll
  for (int v = 0; v < 8; ++v) base[(size_t)(hi * 8 + v) * ld + n] = c[v];
}

// ---------------------------------------------------------------------------
// Simple utility kernels
// ---------------------------------------------------------------------------
__global__ void k_f32_to_f16(const float* __restrict__ in, f16* __restrict__ out, int n) {
  int i = blockIdx.x * blockDim.x + threadIdx.x;
  if (i < n) out[i] = (f16)in[i];
}

__global__ void k_fill(float* __restrict__ p, float v, int n) {
  int i = blockIdx.x * blockDim.x + threadIdx.x;
  if (i < n) p[i] = v;
}

// ---------------------------------------------------------------------------
// Conv2d 16->32, 3x3 SAME on (1024,16,4,8); writes tokens z[(b*32+p)*32+c]
// ---------------------------------------------------------------------------
__global__ void k_conv(const float* __restrict__ x, const float* __restrict__ w,
                       const float* __restrict__ b, float* __restrict__ z,
                       f16* __restrict__ z16) {
  int idx = blockIdx.x * blockDim.x + threadIdx.x;   // 1024*32*32
  int co = idx & 31;
  int p  = (idx >> 5) & 31;
  int bi = idx >> 10;
  int y = p >> 3, xp = p & 7;
  float acc = b[co];
  for (int ci = 0; ci < 16; ++ci) {
    const float* xin = x + ((size_t)(bi * 16 + ci)) * 32;   // 4*8 plane
    const float* wk  = w + ((size_t)(co * 16 + ci)) * 9;
#pragma unroll
    for (int ky = 0; ky < 3; ++ky) {
      int yy = y + ky - 1;
      if (yy < 0 || yy >= 4) continue;
#pragma unroll
      for (int kx = 0; kx < 3; ++kx) {
        int xx = xp + kx - 1;
        if (xx < 0 || xx >= 8) continue;
        acc += xin[yy * 8 + xx] * wk[ky * 3 + kx];
      }
    }
  }
  int t = bi * 32 + p;
  z[(size_t)t * 32 + co]   = acc;
  z16[(size_t)t * 32 + co] = (f16)acc;
}

// ---------------------------------------------------------------------------
// Generic C = A(MxK f16) @ W(NxK f16)^T (+bias), WMMA tiled.
// One wave per 16x16 output tile.
// ---------------------------------------------------------------------------
__global__ void k_gemm_xwT(const f16* __restrict__ A, const f16* __restrict__ W,
                           const float* __restrict__ bias,
                           float* __restrict__ Cf32, f16* __restrict__ Cf16,
                           int N, int K) {
  const int m0 = blockIdx.x * 16;
  const int n0 = blockIdx.y * 16;
  const int lane = threadIdx.x;
  v8f c = {};
  for (int k0 = 0; k0 < K; k0 += 32) {
    v16h a = ld_tile(A + (size_t)m0 * K + k0, K, lane);
    v16h b = ld_tile(W + (size_t)n0 * K + k0, K, lane);
    c = wmma32(a, b, c);
  }
  const int n  = lane & 15;
  const int hi = (lane >> 4) & 1;
  const float bv = bias ? bias[n0 + n] : 0.0f;
#pragma unroll
  for (int v = 0; v < 8; ++v) {
    const int m = hi * 8 + v;
    float val = c[v] + bv;
    if (Cf32) Cf32[(size_t)(m0 + m) * N + n0 + n] = val;
    if (Cf16) Cf16[(size_t)(m0 + m) * N + n0 + n] = (f16)val;
  }
}

// ---------------------------------------------------------------------------
// Fused attention: per (nb, h, m-tile16): QK^T (WMMA) -> exact softmax -> P.V
// qkv16: (S*Nb, 96) f16 tokens t = s*32+nb; q|k|v at +0/+32/+64, head h at h*8.
// LDS: K(16KB) V(16KB) scores(64KB) = 96KB dynamic.
// K/V rows are 16B-aligned (token*192B + h*16B) -> b128 staging; on gfx1250
// use GLOBAL_LOAD_ASYNC_TO_LDS_B128 (ASYNCcnt) to bypass VGPRs.
// ---------------------------------------------------------------------------
#define S_LEN 1024
__global__ void __launch_bounds__(256) k_attn(const f16* __restrict__ qkv,
                                              float* __restrict__ o) {
  const int bid  = blockIdx.x;
  const int mt   = bid & 63;
  const int h    = (bid >> 6) & 3;
  const int nb   = bid >> 8;
  const int tid  = threadIdx.x;
  const int wave = tid >> 5, lane = tid & 31;

  extern __shared__ char smem[];
  f16*   Kl = (f16*)smem;                      // 1024 x 8
  f16*   Vl = Kl + S_LEN * 8;                  // 1024 x 8
  float* Sc = (float*)(Vl + S_LEN * 8);        // 16 x 1024 f32
  float* Pt = (float*)Kl;                      // reused for PV partials (8x256 f32)

  // stage K,V for this (nb,h): one 16B chunk per row per matrix
#if USE_ASYNC_LDS
  for (int s = tid; s < S_LEN; s += 256) {
    const uint4* rowp = (const uint4*)(qkv + ((size_t)(s * 32 + nb)) * 96);
    // global side: flat and AS1 addresses are bit-identical for global memory
    __builtin_amdgcn_global_load_async_to_lds_b128(
        (as1_v4i)(unsigned long long)(rowp + 4 + h),
        (as3_v4i)(Kl + s * 8), 0, 0);
    __builtin_amdgcn_global_load_async_to_lds_b128(
        (as1_v4i)(unsigned long long)(rowp + 8 + h),
        (as3_v4i)(Vl + s * 8), 0, 0);
  }
  __builtin_amdgcn_s_wait_asynccnt(0);
#else
  for (int s = tid; s < S_LEN; s += 256) {
    const uint4* rowp = (const uint4*)(qkv + ((size_t)(s * 32 + nb)) * 96);
    ((uint4*)Kl)[s] = rowp[4 + h];
    ((uint4*)Vl)[s] = rowp[8 + h];
  }
#endif
  __syncthreads();

  // Q tile (16 rows, K=8 padded to 32); row stride between s values = 32*96
  const f16* qbase = qkv + ((size_t)((mt * 16) * 32 + nb)) * 96 + h * 8;
  const v16h aQ = ld_tile_pad(qbase, 32 * 96, lane, 8);

  // scores: 64 N-tiles split over 8 waves
  for (int i = 0; i < 8; ++i) {
    const int nt = wave + i * 8;
    v16h bK = ld_tile_pad(Kl + nt * 16 * 8, 8, lane, 8);
    v8f c = {};
    c = wmma32(aQ, bK, c);
    st_cd(Sc + nt * 16, S_LEN, lane, c);
  }
  __syncthreads();

  // row softmax (scale 1/sqrt(8)); wave handles rows 2w, 2w+1
  const float scale = 0.35355339059327373f;
#pragma unroll
  for (int rr = 0; rr < 2; ++rr) {
    float* srow = Sc + (wave * 2 + rr) * S_LEN;
    float mx = -1e30f;
    for (int j = lane; j < S_LEN; j += 32) mx = fmaxf(mx, srow[j] * scale);
    for (int off = 16; off > 0; off >>= 1) mx = fmaxf(mx, __shfl_xor(mx, off, 32));
    float sum = 0.0f;
    for (int j = lane; j < S_LEN; j += 32) {
      float e = __expf(srow[j] * scale - mx);
      srow[j] = e;
      sum += e;
    }
    for (int off = 16; off > 0; off >>= 1) sum += __shfl_xor(sum, off, 32);
    const float inv = 1.0f / sum;
    for (int j = lane; j < S_LEN; j += 32) srow[j] *= inv;
  }
  __syncthreads();

  // P @ V : K=1024 in 32 steps, 4 per wave
  v8f acc = {};
  for (int i = 0; i < 4; ++i) {
    const int ks = wave + i * 8;
    v16h aP = ld_tile_f32(Sc + ks * 32, S_LEN, lane);
    v16h bV = ld_tileB_v(Vl, lane, ks * 32);
    acc = wmma32(aP, bV, acc);
  }
  st_cd(Pt + wave * 256, 16, lane, acc);
  __syncthreads();

  // cross-wave reduce; only columns < HEAD_D valid
  const int mm = tid >> 4, nn = tid & 15;
  float v = 0.0f;
#pragma unroll
  for (int w2 = 0; w2 < 8; ++w2) v += Pt[w2 * 256 + mm * 16 + nn];
  if (nn < 8) {
    const int t = (mt * 16 + mm) * 32 + nb;
    o[(size_t)t * 32 + h * 8 + nn] = v;
  }
}

// ---------------------------------------------------------------------------
// out_proj + residual + LayerNorm1 (one wave per token row, lane = channel)
// ---------------------------------------------------------------------------
__global__ void __launch_bounds__(256) k_outproj_ln(
    const float* __restrict__ z, const float* __restrict__ o,
    const float* __restrict__ Wp, const float* __restrict__ bp,
    const float* __restrict__ lnw, const float* __restrict__ lnb,
    float* __restrict__ z1, f16* __restrict__ z1h) {
  const int row  = blockIdx.x * 8 + (threadIdx.x >> 5);
  const int lane = threadIdx.x & 31;
  const float ov = o[(size_t)row * 32 + lane];
  float acc = bp[lane];
#pragma unroll
  for (int k = 0; k < 32; ++k) {
    acc += __shfl(ov, k, 32) * Wp[lane * 32 + k];
  }
  float r = z[(size_t)row * 32 + lane] + acc;
  float m = r;
  for (int off = 16; off > 0; off >>= 1) m += __shfl_xor(m, off, 32);
  m *= (1.0f / 32.0f);
  float d = r - m;
  float var = d * d;
  for (int off = 16; off > 0; off >>= 1) var += __shfl_xor(var, off, 32);
  var *= (1.0f / 32.0f);
  float out = d * rsqrtf(var + 1e-5f) * lnw[lane] + lnb[lane];
  z1[(size_t)row * 32 + lane]  = out;
  z1h[(size_t)row * 32 + lane] = (f16)out;
}

// ---------------------------------------------------------------------------
// Fused FF1(32->2048)+ReLU+FF2(2048->32)+residual+LN2, writes transposed g16.
// 16-token tile/block; hidden tile f16 in LDS (64KB). 8 waves, WMMA throughout.
// ---------------------------------------------------------------------------
__global__ void __launch_bounds__(256) k_ff_ln2(
    const f16* __restrict__ z1h, const float* __restrict__ z1,
    const f16* __restrict__ w1h, const float* __restrict__ b1,
    const f16* __restrict__ w2h, const float* __restrict__ b2,
    const float* __restrict__ lnw, const float* __restrict__ lnb,
    f16* __restrict__ g16) {
  const int t0   = blockIdx.x * 16;
  const int wave = threadIdx.x >> 5, lane = threadIdx.x & 31;
  extern __shared__ char smem[];
  f16*   Hl = (f16*)smem;                  // 16 x 2048 f16  (64KB)
  float* Fo = (float*)(Hl + 16 * 2048);    // 16 x 32 f32
  float* Pt = Fo + 16 * 32;                // 8 x 256 f32 partials

  // FF1: A (16x32) reused for all 128 N-tiles; wave handles 16 of them
  const v16h a1 = ld_tile(z1h + (size_t)t0 * 32, 32, lane);
  for (int i = 0; i < 16; ++i) {
    const int ft = wave * 16 + i;          // hidden-tile index 0..127
    v16h b = ld_tile(w1h + (size_t)(ft * 16) * 32, 32, lane);
    v8f c = {};
    c = wmma32(a1, b, c);
    const int n = lane & 15, hi = (lane >> 4) & 1;
    const float bv = b1[ft * 16 + n];
#pragma unroll
    for (int v = 0; v < 8; ++v) {
      const int m = hi * 8 + v;
      Hl[m * 2048 + ft * 16 + n] = (f16)fmaxf(c[v] + bv, 0.0f);
    }
  }
  __syncthreads();

  // FF2: wave = kc*2 + ntile; each does 16 of 64 K-steps
  const int nt = wave & 1, kc = wave >> 1;
  v8f acc = {};
  for (int i = 0; i < 16; ++i) {
    const int k0 = (kc * 16 + i) * 32;
    v16h aH = ld_tile(Hl + k0, 2048, lane);
    v16h bW = ld_tile(w2h + (size_t)(nt * 16) * 2048 + k0, 2048, lane);
    acc = wmma32(aH, bW, acc);
  }
  st_cd(Pt + wave * 256, 16, lane, acc);
  __syncthreads();

  // reduce K-partials (+bias) into Fo (16x32)
#pragma unroll
  for (int r2 = 0; r2 < 2; ++r2) {
    const int e = threadIdx.x + r2 * 256;
    const int m = e >> 5, n = e & 31;
    const int nt2 = n >> 4, nn = n & 15;
    float s = b2[n];
#pragma unroll
    for (int kc2 = 0; kc2 < 4; ++kc2) s += Pt[(kc2 * 2 + nt2) * 256 + m * 16 + nn];
    Fo[m * 32 + n] = s;
  }
  __syncthreads();

  // residual + LN2, then transpose-write g[(b*32+c)*32+p] = z2[b][p][c]
#pragma unroll
  for (int rr = 0; rr < 2; ++rr) {
    const int m = wave * 2 + rr;
    const int t = t0 + m;
    float r = z1[(size_t)t * 32 + lane] + Fo[m * 32 + lane];
    float mean = r;
    for (int off = 16; off > 0; off >>= 1) mean += __shfl_xor(mean, off, 32);
    mean *= (1.0f / 32.0f);
    float d = r - mean;
    float var = d * d;
    for (int off = 16; off > 0; off >>= 1) var += __shfl_xor(var, off, 32);
    var *= (1.0f / 32.0f);
    float out = d * rsqrtf(var + 1e-5f) * lnw[lane] + lnb[lane];
    const int b = t >> 5, p = t & 31;
    g16[(size_t)(b * 32 + lane) * 32 + p] = (f16)out;
  }
}

// ---------------------------------------------------------------------------
// GCN pieces
// ---------------------------------------------------------------------------
__global__ void k_deg(const int* __restrict__ col, float* __restrict__ deg, int E) {
  int e = blockIdx.x * blockDim.x + threadIdx.x;
  if (e < E) atomicAdd(&deg[col[e]], 1.0f);
}

__global__ void k_rsqrt_ip(float* __restrict__ d, int n) {
  int i = blockIdx.x * blockDim.x + threadIdx.x;
  if (i < n) d[i] = rsqrtf(d[i]);
}

// self-loop term (norm = 1/deg) + bias
__global__ void k_gcn_self(const float* __restrict__ xw, const float* __restrict__ dis,
                           const float* __restrict__ bias, float* __restrict__ out,
                           int nelem) {
  int i = blockIdx.x * blockDim.x + threadIdx.x;
  if (i >= nelem) return;
  const int node = i >> 5, f = i & 31;
  const float dn = dis[node];
  out[i] = xw[i] * dn * dn + bias[f];
}

// edge scatter: one wave per edge, lane per feature (coalesced 128B gather +
// coalesced 128B of f32 atomics). HBM-bound stage (~1.1 GB total traffic).
__global__ void k_gcn_scatter(const int* __restrict__ row, const int* __restrict__ col,
                              const float* __restrict__ xw, const float* __restrict__ dis,
                              float* __restrict__ out, int E) {
  const long long tid = (long long)blockIdx.x * blockDim.x + threadIdx.x;
  const int e = (int)(tid >> 5);
  const int f = (int)(tid & 31);
  if (e >= E) return;
  const int r = row[e], c = col[e];
  const float nrm = dis[r] * dis[c];
  atomicAdd(&out[(size_t)c * 32 + f], xw[(size_t)r * 32 + f] * nrm);
}

__global__ void k_relu_f16(const float* __restrict__ in, f16* __restrict__ out, int n) {
  int i = blockIdx.x * blockDim.x + threadIdx.x;
  if (i < n) out[i] = (f16)fmaxf(in[i], 0.0f);
}

// ---------------------------------------------------------------------------
// FC (32->10) + log_softmax, one thread per node
// ---------------------------------------------------------------------------
__global__ void k_fc_lsm(const float* __restrict__ g, const float* __restrict__ W,
                         const float* __restrict__ b, float* __restrict__ out, int n) {
  int i = blockIdx.x * blockDim.x + threadIdx.x;
  if (i >= n) return;
  const float* x = g + (size_t)i * 32;
  float logits[10];
  float mx = -1e30f;
#pragma unroll
  for (int c = 0; c < 10; ++c) {
    float acc = b[c];
    const float* w = W + c * 32;
#pragma unroll
    for (int k = 0; k < 32; ++k) acc += x[k] * w[k];
    logits[c] = acc;
    mx = fmaxf(mx, acc);
  }
  float sum = 0.0f;
#pragma unroll
  for (int c = 0; c < 10; ++c) sum += __expf(logits[c] - mx);
  const float lse = mx + __logf(sum);
#pragma unroll
  for (int c = 0; c < 10; ++c) out[(size_t)i * 10 + c] = logits[c] - lse;
}

// ---------------------------------------------------------------------------
// Host-side orchestration (graph-capture safe: launches only, all on `stream`)
// ---------------------------------------------------------------------------
extern "C" void kernel_launch(void* const* d_in, const int* in_sizes, int n_in,
                              void* d_out, int out_size, void* d_ws, size_t ws_size,
                              hipStream_t stream) {
  const float* x          = (const float*)d_in[0];
  const int*   edge_index = (const int*)d_in[1];
  const float* conv_w     = (const float*)d_in[2];
  const float* conv_b     = (const float*)d_in[3];
  const float* in_proj_w  = (const float*)d_in[4];
  const float* in_proj_b  = (const float*)d_in[5];
  const float* out_proj_w = (const float*)d_in[6];
  const float* out_proj_b = (const float*)d_in[7];
  const float* ln1_w      = (const float*)d_in[8];
  const float* ln1_b      = (const float*)d_in[9];
  const float* ln2_w      = (const float*)d_in[10];
  const float* ln2_b      = (const float*)d_in[11];
  const float* ff1_w      = (const float*)d_in[12];
  const float* ff1_b      = (const float*)d_in[13];
  const float* ff2_w      = (const float*)d_in[14];
  const float* ff2_b      = (const float*)d_in[15];
  const float* gcn_w      = (const float*)d_in[16];
  const float* gcn_b      = (const float*)d_in[17];
  const float* fc_w       = (const float*)d_in[18];
  const float* fc_b       = (const float*)d_in[19];

  const int E    = in_sizes[1] / 2;      // 2097152
  const int NTOK = 32768;                // tokens == graph nodes
  const int* row = edge_index;
  const int* col = edge_index + E;

  // workspace carve-out
  char* base = (char*)d_ws;
  size_t off = 0;
  auto carve = [&](size_t bytes) -> char* {
    char* p = base + off;
    off = (off + bytes + 255) & ~(size_t)255;
    return p;
  };
  float* z      = (float*)carve((size_t)NTOK * 32 * 4);
  f16*   z16    = (f16*)  carve((size_t)NTOK * 32 * 2);
  f16*   qkv16  = (f16*)  carve((size_t)NTOK * 96 * 2);
  float* o      = (float*)carve((size_t)NTOK * 32 * 4);
  float* z1     = (float*)carve((size_t)NTOK * 32 * 4);
  f16*   z1h    = (f16*)  carve((size_t)NTOK * 32 * 2);
  f16*   g16    = (f16*)  carve((size_t)NTOK * 32 * 2);
  float* xw     = (float*)carve((size_t)NTOK * 32 * 4);
  float* out1   = (float*)carve((size_t)NTOK * 32 * 4);
  f16*   g2h    = (f16*)  carve((size_t)NTOK * 32 * 2);
  float* out2   = (float*)carve((size_t)NTOK * 32 * 4);
  float* dis    = (float*)carve((size_t)NTOK * 4);
  f16*   iph    = (f16*)  carve(96 * 32 * 2);
  f16*   w1h    = (f16*)  carve(2048 * 32 * 2);
  f16*   w2h    = (f16*)  carve(32 * 2048 * 2);
  f16*   gwh    = (f16*)  carve(32 * 32 * 2);
  (void)ws_size; (void)n_in; (void)out_size;

  // 0) weight conversions to f16
  k_f32_to_f16<<<(96 * 32 + 255) / 256, 256, 0, stream>>>(in_proj_w, iph, 96 * 32);
  k_f32_to_f16<<<(2048 * 32 + 255) / 256, 256, 0, stream>>>(ff1_w, w1h, 2048 * 32);
  k_f32_to_f16<<<(32 * 2048 + 255) / 256, 256, 0, stream>>>(ff2_w, w2h, 32 * 2048);
  k_f32_to_f16<<<(32 * 32 + 255) / 256, 256, 0, stream>>>(gcn_w, gwh, 32 * 32);

  // 1) conv -> z (f32 + f16)
  k_conv<<<(NTOK * 32) / 256, 256, 0, stream>>>(x, conv_w, conv_b, z, z16);

  // 2) qkv = z @ in_proj_w^T + b   (M=32768, N=96, K=32) -> f16
  k_gemm_xwT<<<dim3(NTOK / 16, 96 / 16), 32, 0, stream>>>(
      z16, iph, in_proj_b, nullptr, qkv16, 96, 32);

  // 3) fused attention -> o  (32 batches x 4 heads x 64 m-tiles)
  k_attn<<<32 * 4 * 64, 256, 96 * 1024, stream>>>(qkv16, o);

  // 4) out_proj + residual + LN1 -> z1 (f32 + f16)
  k_outproj_ln<<<NTOK / 8, 256, 0, stream>>>(z, o, out_proj_w, out_proj_b,
                                             ln1_w, ln1_b, z1, z1h);

  // 5) fused FF + residual + LN2 -> transposed g16
  k_ff_ln2<<<NTOK / 16, 256, 16 * 2048 * 2 + 16 * 32 * 4 + 8 * 256 * 4, stream>>>(
      z1h, z1, w1h, ff1_b, w2h, ff2_b, ln2_w, ln2_b, g16);

  // 6) degree (self-loops=1) -> dis = rsqrt(deg)
  k_fill<<<(NTOK + 255) / 256, 256, 0, stream>>>(dis, 1.0f, NTOK);
  k_deg<<<(E + 255) / 256, 256, 0, stream>>>(col, dis, E);
  k_rsqrt_ip<<<(NTOK + 255) / 256, 256, 0, stream>>>(dis, NTOK);

  const int nelem = NTOK * 32;
  const long long scat = (long long)E * 32;
  const int scat_blocks = (int)((scat + 255) / 256);

  // 7) GCN layer 1
  k_gemm_xwT<<<dim3(NTOK / 16, 2), 32, 0, stream>>>(g16, gwh, nullptr, xw, nullptr, 32, 32);
  k_gcn_self<<<nelem / 256, 256, 0, stream>>>(xw, dis, gcn_b, out1, nelem);
  k_gcn_scatter<<<scat_blocks, 256, 0, stream>>>(row, col, xw, dis, out1, E);
  k_relu_f16<<<nelem / 256, 256, 0, stream>>>(out1, g2h, nelem);

  // 8) GCN layer 2 (shared weights)
  k_gemm_xwT<<<dim3(NTOK / 16, 2), 32, 0, stream>>>(g2h, gwh, nullptr, xw, nullptr, 32, 32);
  k_gcn_self<<<nelem / 256, 256, 0, stream>>>(xw, dis, gcn_b, out2, nelem);
  k_gcn_scatter<<<scat_blocks, 256, 0, stream>>>(row, col, xw, dis, out2, E);

  // 9) FC + log_softmax -> d_out (32768 x 10 f32)
  k_fc_lsm<<<(NTOK + 255) / 256, 256, 0, stream>>>(out2, fc_w, fc_b, (float*)d_out, NTOK);
}
